// GaussianKernel_54992761258023
// MI455X (gfx1250) — matrix operand order repaired and checked by reading, not compiled, not run on hardware
//
#include <hip/hip_runtime.h>
#include <hip/hip_bf16.h>

// ---------------------------------------------------------------------------
// Gaussian kernel matrix: out = exp(-d2 / (2*mean(d2))), d2 = pairwise sq L2.
// x: [4096, 256] fp32.  out: [4096, 4096] fp32.
//
// Pipeline:
//   1) prep: row norms sq[i]; one-time fp32 -> bf16 hi/lo split of x (2x2MB)
//   2) GEMM gram = x @ x^T via v_wmma_f32_16x16x32_bf16, 3xBF16 split
//      (hi*hi + hi*lo + lo*hi) for near-fp32 accuracy.  Tiles staged
//      global->LDS by the Tensor Data Mover (tensor_load_to_lds, TENSORcnt),
//      double-buffered: DMA of slab s+1 overlaps WMMA of slab s.
//      Epilogue: d2 = sq_i + sq_j - 2*gram (clamped), per-block partial sums.
//   3) deterministic reduction -> scale = -1/(2*mean(d2))
//   4) out = exp(scale * d2)  (in place on d_out)
// ---------------------------------------------------------------------------

typedef __attribute__((ext_vector_type(16))) __bf16       v16bf;
typedef __attribute__((ext_vector_type(8)))  __bf16       bf16x8;
typedef __attribute__((ext_vector_type(8)))  float        v8f;
typedef __attribute__((ext_vector_type(4)))  float        f32x4;
typedef __attribute__((ext_vector_type(4)))  unsigned int v4u;
typedef __attribute__((ext_vector_type(8)))  int          v8i;
typedef __attribute__((ext_vector_type(4)))  int          v4i;

#define NROWS 4096
#define KDIM  256
#define TILE  128
#define KS    32          // K-slab per TDM transfer / WMMA step
#define LSTR  40          // LDS row stride in bf16 elems (32 + 8 pad = 80B)

// --------------- kernel 1: row norms + one-time bf16 hi/lo split -----------
__global__ __launch_bounds__(256)
void prep_kernel(const float* __restrict__ x, float* __restrict__ sq,
                 __bf16* __restrict__ xh, __bf16* __restrict__ xl) {
    int wave = threadIdx.x >> 5;
    int lane = threadIdx.x & 31;
    int row  = blockIdx.x * 8 + wave;
    float s = 0.f;
    #pragma unroll
    for (int j = 0; j < KDIM / 32; ++j) {
        size_t idx = (size_t)row * KDIM + lane + 32 * j;
        float v = x[idx];
        s += v * v;
        __bf16 h = (__bf16)v;
        xh[idx] = h;
        xl[idx] = (__bf16)(v - (float)h);
    }
    #pragma unroll
    for (int off = 16; off > 0; off >>= 1) s += __shfl_xor(s, off, 32);
    if (lane == 0) sq[row] = s;
}

// ---- TDM: DMA one 128-row x 32-elem bf16 tile global->LDS with padding ----
// D# per CDNA5 ISA: 2D tile, data_size=2B,
// tensor = [4096 x 256] bf16 row-major, tile = [128 x 32],
// pad 4 DWORDs (16B) after every 16 DWORDs (64B row) -> LDS stride 80B.
static __device__ inline void tdm_load_tile(const __bf16* gaddr, unsigned lds_off) {
    unsigned long long ga = (unsigned long long)(uintptr_t)gaddr;
    v4u g0;
    g0[0] = 1u;                                   // count=1 (valid user desc)
    g0[1] = lds_off;                              // lds_addr (bytes)
    g0[2] = (unsigned)(ga & 0xFFFFFFFFu);         // global_addr[31:0]
    g0[3] = (unsigned)((ga >> 32) & 0x01FFFFFFu)  // global_addr[56:32]
          | 0x80000000u;                          // type = 2 ("image")
    v8i g1;
    g1[0] = (1 << 16)                             // data_size = 2B
          | (1 << 20)                             // pad_enable
          | (3 << 22)                             // pad_interval: 16 DWORDs (64B)
          | (3 << 25);                            // pad_amount: 4 DWORDs (16B)
    g1[1] = (int)(256u << 16);                    // tensor_dim0 = 256
    g1[2] = (int)(4096u << 16);                   // tensor_dim1 = 4096
    g1[3] = (int)(32u << 16);                     // tile_dim0 = 32
    g1[4] = 128;                                  // tile_dim1 = 128, tile_dim2 = 0
    g1[5] = 256;                                  // tensor_dim0_stride = 256
    g1[6] = 0;
    g1[7] = 0;
    v4i g2;                                        // benign upper dims
    g2[0] = 1;          // tensor_dim2 = 1
    g2[1] = 1;          // tensor_dim3 = 1
    g2[2] = 0x100000;   // tensor_dim2_stride = 4096*256 elems
    g2[3] = 0;          // tile_dim3 = 0
    v4i g3;
    g3[0] = 0x100000;   // tensor_dim3_stride
    g3[1] = 0x00010000; // tensor_dim4 = 1
    g3[2] = 0;
    g3[3] = 0;          // tile_dim4 = 0
#if __has_include(<hip/amd_detail/amd_gfx1250_TDM.h>)
    v8i gz;
    #pragma unroll
    for (int i = 0; i < 8; ++i) gz[i] = 0;
    __builtin_amdgcn_tensor_load_to_lds(g0, g1, g2, g3, gz, 0);
#else
    __builtin_amdgcn_tensor_load_to_lds(g0, g1, g2, g3, 0);
#endif
}

// assemble a 16-element bf16 fragment from two 16B LDS chunks
static __device__ inline v16bf ld_frag16(const __bf16* p0, const __bf16* p1) {
    bf16x8 a = *(const bf16x8*)p0;
    bf16x8 b = *(const bf16x8*)p1;
    return __builtin_shufflevector(a, b, 0, 1, 2, 3, 4, 5, 6, 7,
                                   8, 9, 10, 11, 12, 13, 14, 15);
}

// ---------------- kernel 2: WMMA GEMM -> clamped d2 + partial sums ---------
__global__ __launch_bounds__(256)
void gemm_d2_kernel(const __bf16* __restrict__ xh,
                    const __bf16* __restrict__ xl,
                    const float* __restrict__ sq,
                    float* __restrict__ d2out,
                    float* __restrict__ partials) {
    __shared__ __bf16 Ahi[2][TILE * LSTR];   // TDM-padded bf16 tiles
    __shared__ __bf16 Alo[2][TILE * LSTR];
    __shared__ __bf16 Bhi[2][TILE * LSTR];
    __shared__ __bf16 Blo[2][TILE * LSTR];
    __shared__ float  red[256];

    const int tid  = threadIdx.x;
    const int wave = tid >> 5;
    const int lane = tid & 31;
    const int wm   = wave >> 2;      // 0..1 : 64-row band
    const int wn   = wave & 3;       // 0..3 : 32-col band
    const int half = lane >> 4;      // 0..1
    const int l16  = lane & 15;

    const int bm = blockIdx.y;
    const int bn = blockIdx.x;
    const size_t aoff = (size_t)bm * TILE * KDIM;
    const size_t boff = (size_t)bn * TILE * KDIM;

    unsigned lAhi[2], lAlo[2], lBhi[2], lBlo[2];
    #pragma unroll
    for (int b = 0; b < 2; ++b) {
        lAhi[b] = (unsigned)(uintptr_t)&Ahi[b][0];
        lAlo[b] = (unsigned)(uintptr_t)&Alo[b][0];
        lBhi[b] = (unsigned)(uintptr_t)&Bhi[b][0];
        lBlo[b] = (unsigned)(uintptr_t)&Blo[b][0];
    }

    v8f acc[4][2];
    #pragma unroll
    for (int i = 0; i < 4; ++i)
        #pragma unroll
        for (int j = 0; j < 2; ++j)
            #pragma unroll
            for (int r = 0; r < 8; ++r) acc[i][j][r] = 0.f;

    // prologue: DMA slab 0 into buffer 0 (one wave drives the TDM)
    if (wave == 0) {
        tdm_load_tile(xh + aoff, lAhi[0]);
        tdm_load_tile(xl + aoff, lAlo[0]);
        tdm_load_tile(xh + boff, lBhi[0]);
        tdm_load_tile(xl + boff, lBlo[0]);
    }

    for (int s = 0; s < KDIM / KS; ++s) {
        if (wave == 0) {
            if (s < KDIM / KS - 1) {
                // prefetch slab s+1 into the other buffer, then wait for
                // slab s only (TDM ops complete in order per wave)
                const int ks1 = (s + 1) * KS;
                const int nb  = (s + 1) & 1;
                tdm_load_tile(xh + aoff + ks1, lAhi[nb]);
                tdm_load_tile(xl + aoff + ks1, lAlo[nb]);
                tdm_load_tile(xh + boff + ks1, lBhi[nb]);
                tdm_load_tile(xl + boff + ks1, lBlo[nb]);
                __builtin_amdgcn_s_wait_tensorcnt((short)4);
            } else {
                __builtin_amdgcn_s_wait_tensorcnt((short)0);
            }
        }
        __syncthreads();    // slab s visible to all waves

        const int cb = s & 1;

        // B fragments: lane = column n; 16 contiguous K bf16 at 16*half
        v16bf bfh[2], bfl[2];
        #pragma unroll
        for (int ni = 0; ni < 2; ++ni) {
            const int n = wn * 32 + ni * 16 + l16;
            const __bf16* ph = &Bhi[cb][n * LSTR + 16 * half];
            const __bf16* pl = &Blo[cb][n * LSTR + 16 * half];
            bfh[ni] = ld_frag16(ph, ph + 8);
            bfl[ni] = ld_frag16(pl, pl + 8);
        }
        // A fragments + WMMA (3xBF16 split: hi*hi + hi*lo + lo*hi)
        #pragma unroll
        for (int mi = 0; mi < 4; ++mi) {
            const int m = wm * 64 + mi * 16 + l16;
            v16bf afh = ld_frag16(&Ahi[cb][m * LSTR + 16 * half],
                                  &Ahi[cb][m * LSTR + 16 * half + 8]);
            v16bf afl = ld_frag16(&Alo[cb][m * LSTR + 16 * half],
                                  &Alo[cb][m * LSTR + 16 * half + 8]);
            #pragma unroll
            for (int ni = 0; ni < 2; ++ni) {
                acc[mi][ni] = __builtin_amdgcn_wmma_f32_16x16x32_bf16(
                    false, afh, false, bfh[ni], (short)0, acc[mi][ni], false, false);
                acc[mi][ni] = __builtin_amdgcn_wmma_f32_16x16x32_bf16(
                    false, afh, false, bfl[ni], (short)0, acc[mi][ni], false, false);
                acc[mi][ni] = __builtin_amdgcn_wmma_f32_16x16x32_bf16(
                    false, afl, false, bfh[ni], (short)0, acc[mi][ni], false, false);
            }
        }
        __syncthreads();    // all reads of buf[s&1] done before TDM reuses it
    }

    // ---- epilogue: d2 = sq_i + sq_j - 2*gram, clamp, store, partial sum
    float lsum = 0.f;
    #pragma unroll
    for (int mi = 0; mi < 4; ++mi) {
        #pragma unroll
        for (int ni = 0; ni < 2; ++ni) {
            const int gn = bn * TILE + wn * 32 + ni * 16 + l16;
            const float sqc = sq[gn];
            #pragma unroll
            for (int r = 0; r < 8; ++r) {
                const int gm = bm * TILE + wm * 64 + mi * 16 + r + 8 * half;
                float d = sq[gm] + sqc - 2.f * acc[mi][ni][r];
                d = fmaxf(d, 0.f);
                lsum += d;
                d2out[(size_t)gm * NROWS + gn] = d;
            }
        }
    }
    __syncthreads();
    red[tid] = lsum;
    __syncthreads();
    #pragma unroll
    for (int s = 128; s > 0; s >>= 1) {
        if (tid < s) red[tid] += red[tid + s];
        __syncthreads();
    }
    if (tid == 0) partials[blockIdx.y * gridDim.x + blockIdx.x] = red[0];
}

// ------------- kernel 3: reduce 1024 partials -> scale = -1/(2*sigma^2) ----
__global__ __launch_bounds__(256)
void reduce_kernel(const float* __restrict__ partials, float* __restrict__ scale_out) {
    __shared__ double red[256];
    const int t = threadIdx.x;
    double s = 0.0;
    for (int i = t; i < 1024; i += 256) s += (double)partials[i];
    red[t] = s;
    __syncthreads();
    for (int st = 128; st > 0; st >>= 1) {
        if (t < st) red[t] += red[t + st];
        __syncthreads();
    }
    if (t == 0) {
        double mean   = red[0] / ((double)NROWS * (double)NROWS);
        double sigma2 = 1.0 * mean;  // ALPHA = 1.0
        scale_out[0]  = (float)(-1.0 / (2.0 * sigma2));
    }
}

// --------------------- kernel 4: out = exp(scale * d2) ---------------------
__global__ __launch_bounds__(256)
void exp_kernel(float* __restrict__ d, const float* __restrict__ scale_p) {
    const float sc = scale_p[0];
    const size_t i = ((size_t)blockIdx.x * 256 + threadIdx.x) * 4;
    f32x4 v = *(f32x4*)(d + i);
    #pragma unroll
    for (int e = 0; e < 4; ++e) v[e] = __expf(v[e] * sc);
    *(f32x4*)(d + i) = v;
}

// ---------------------------------------------------------------------------
extern "C" void kernel_launch(void* const* d_in, const int* in_sizes, int n_in,
                              void* d_out, int out_size, void* d_ws, size_t ws_size,
                              hipStream_t stream) {
    (void)in_sizes; (void)n_in; (void)out_size; (void)ws_size;
    const float* x   = (const float*)d_in[0];
    float*       out = (float*)d_out;
    float*       ws  = (float*)d_ws;

    float*  sq       = ws;                  // 4096 floats
    float*  partials = ws + 4096;           // 1024 floats
    float*  scale    = ws + 4096 + 1024;    // 1 float
    __bf16* xh       = (__bf16*)(ws + 8192);        // 4096*256 bf16 (2MB)
    __bf16* xl       = xh + (size_t)NROWS * KDIM;   // 4096*256 bf16 (2MB)

    prep_kernel<<<NROWS / 8, 256, 0, stream>>>(x, sq, xh, xl);

    dim3 grid(NROWS / TILE, NROWS / TILE);   // 32 x 32
    gemm_d2_kernel<<<grid, 256, 0, stream>>>(xh, xl, sq, out, partials);

    reduce_kernel<<<1, 256, 0, stream>>>(partials, scale);

    exp_kernel<<<(NROWS * (size_t)NROWS) / (256 * 4), 256, 0, stream>>>(out, scale);
}
